// Gpt_33552284516545
// MI455X (gfx1250) — matrix operand order repaired and checked by reading, not compiled, not run on hardware
//
#include <hip/hip_runtime.h>
#include <hip/hip_bf16.h>
#include <cmath>
#include <cstdint>

typedef unsigned int   u32;
typedef unsigned short u16;
typedef __bf16 v16bf __attribute__((ext_vector_type(16)));
typedef float  v8f   __attribute__((ext_vector_type(8)));
typedef int    v4i   __attribute__((vector_size(16)));

// Problem constants (match reference)
enum : int { GV = 50257, GD = 768, GH = 12, GDK = 64, GL = 4, GB = 2, GT = 1024, GFF = 3072, NT = GB * GT };

#define TM 128
#define TN 64
#define TK 64          // K-tile (2 WMMA K-steps); all K in model are multiples of 64

// ---- CDNA5 async global->LDS (ASYNCcnt) with portable fallback ------------
#if defined(__has_builtin)
#  if __has_builtin(__builtin_amdgcn_global_load_async_to_lds_b128) && \
      __has_builtin(__builtin_amdgcn_s_wait_asynccnt)
#    define HAS_ASYNC_LDS 1
#  endif
#endif
#ifndef HAS_ASYNC_LDS
#  define HAS_ASYNC_LDS 0
#endif

typedef __attribute__((address_space(1))) v4i* gptr4;   // global (AS1)
typedef __attribute__((address_space(3))) v4i* lptr4;   // LDS (AS3)

__device__ inline void async_copy_b128(const void* gsrc, void* lds_dst) {
#if HAS_ASYNC_LDS
  __builtin_amdgcn_global_load_async_to_lds_b128(
      (gptr4)(uintptr_t)gsrc,
      (lptr4)(u32)(uintptr_t)lds_dst,   // low 32 bits of generic LDS ptr = LDS offset
      0, 0);
#else
  *(uint4*)lds_dst = *(const uint4*)gsrc;
#endif
}
__device__ inline void async_wait() {
#if HAS_ASYNC_LDS
  __builtin_amdgcn_s_wait_asynccnt(0);
#endif
}

__device__ inline u16 f2bf(float f) {
  u32 u = __builtin_bit_cast(u32, f);
  u += 0x7FFFu + ((u >> 16) & 1u);      // round-to-nearest-even
  return (u16)(u >> 16);
}

union Frag { u32 u[8]; v16bf v; };

// LDS tiles: row stride = 32 dwords (64 bf16) = 8 chunks of 16B.
// XOR-swizzle the chunk index by (row & 7): every fragment access stays a 16B
// ds_load_b128 while banks decorrelate across rows.
__device__ inline u32* lds_chunk(u32* base, int row, int chunk) {
  return base + row * 32 + ((chunk ^ (row & 7)) << 2);
}
__device__ inline const u32* lds_chunk(const u32* base, int row, int chunk) {
  return base + row * 32 + ((chunk ^ (row & 7)) << 2);
}

// ---------------------------------------------------------------------------
// bf16 WMMA GEMM: C = epilogue(A[MxK] * B^T + bias), B given [N][K] bf16.
//   All rows 16B-aligned; M multiple of TM; K multiple of TK.
//   Ragged N handled by clamping the staged B row + store guard.
//   Double-buffered LDS, async global->LDS staging (ASYNCcnt).
//   Batch over blockIdx.z: off = (z/ZH)*s1 + (z%ZH)*s2 for A, B, C.
//   EPI: 0 fp32 (+bias) | 1 bf16 ((acc+bias)*scale) | 2 bf16 gelu(acc+bias)
//        3 fp32 acc+bias+residual | 4 bf16 head-transposed store (for V):
//          Cb[(row>>10)*GD*GT + col*GT + (row&1023)]
// ---------------------------------------------------------------------------
template <int EPI>
__global__ __launch_bounds__(256) void wmma_gemm(
    const u16* __restrict__ A, int lda, long sA1, long sA2,
    const u16* __restrict__ B, int ldb, long sB1, long sB2,
    int M, int N, int K,
    const float* __restrict__ bias,
    const float* __restrict__ residual, int ldres,
    float* __restrict__ Cf, u16* __restrict__ Cb, int ldc, long sC1, long sC2,
    float scale, int ZH)
{
  __shared__ __align__(16) u16 As[2][TM * TK];   // 2 x 16 KB
  __shared__ __align__(16) u16 Bs[2][TN * TK];   // 2 x  8 KB

  const int z  = blockIdx.z;
  const int zq = z / ZH, zr = z % ZH;
  const u16* Ab = A + (long)zq * sA1 + (long)zr * sA2;
  const u16* Bb = B + (long)zq * sB1 + (long)zr * sB2;
  const long coff = (long)zq * sC1 + (long)zr * sC2;

  const int m0   = blockIdx.y * TM;
  const int n0   = blockIdx.x * TN;
  const int tid  = threadIdx.x;
  const int lane = tid & 31;
  const int wave = tid >> 5;
  const int wm   = wave >> 1;      // 0..3 -> 32-row strip of M
  const int wn   = wave & 1;       // 0..1 -> 32-col strip of N

  // staging coordinates
  const int arow = tid >> 1;             // 0..127, 4 chunks each
  const int acnk = (tid & 1) * 4;
  const int brow = tid >> 2;             // 0..63, 2 chunks each
  const int bcnk = (tid & 3) * 2;
  int bn = n0 + brow; if (bn > N - 1) bn = N - 1;   // clamp ragged N (LM head)

  auto stage = [&](int k0, int buf) {
    const u16* sa = Ab + (long)(m0 + arow) * lda + k0 + acnk * 8;
    u32* da = (u32*)As[buf];
#pragma unroll
    for (int i = 0; i < 4; ++i)
      async_copy_b128(sa + i * 8, lds_chunk(da, arow, acnk + i));
    const u16* sb = Bb + (long)bn * ldb + k0 + bcnk * 8;
    u32* db = (u32*)Bs[buf];
#pragma unroll
    for (int i = 0; i < 2; ++i)
      async_copy_b128(sb + i * 8, lds_chunk(db, brow, bcnk + i));
  };

  v8f acc[2][2] = {};

  stage(0, 0);
  async_wait();
  __syncthreads();

  const int r16 = lane & 15;
  const int cA  = (lane < 16) ? 0 : 1;   // A: lanes 16-31 start at K=8 -> chunk +1
  const int cB  = (lane < 16) ? 0 : 1;   // B: lanes 16-31 start at K=8 -> chunk +1

  int cur = 0;
  for (int k0 = 0; k0 < K; k0 += TK) {
    if (k0 + TK < K) stage(k0 + TK, cur ^ 1);   // prefetch next tile (async)

    const u32* Acur = (const u32*)As[cur];
    const u32* Bcur = (const u32*)Bs[cur];
#pragma unroll
    for (int ks = 0; ks < 2; ++ks) {            // 2 K-sub-steps of 32
      Frag af[2], bfr[2];
#pragma unroll
      for (int mi = 0; mi < 2; ++mi) {
        const int row = wm * 32 + mi * 16 + r16;
        const u32* p0 = lds_chunk(Acur, row, ks * 4 + cA);
        const u32* p1 = lds_chunk(Acur, row, ks * 4 + cA + 2);
#pragma unroll
        for (int vv = 0; vv < 4; ++vv) { af[mi].u[vv] = p0[vv]; af[mi].u[4 + vv] = p1[vv]; }
      }
#pragma unroll
      for (int ni = 0; ni < 2; ++ni) {
        const int row = wn * 32 + ni * 16 + r16;
        const u32* q0 = lds_chunk(Bcur, row, ks * 4 + cB);
        const u32* q1 = lds_chunk(Bcur, row, ks * 4 + cB + 2);
#pragma unroll
        for (int vv = 0; vv < 4; ++vv) { bfr[ni].u[vv] = q0[vv]; bfr[ni].u[4 + vv] = q1[vv]; }
      }
#pragma unroll
      for (int mi = 0; mi < 2; ++mi)
#pragma unroll
        for (int ni = 0; ni < 2; ++ni)
          acc[mi][ni] = __builtin_amdgcn_wmma_f32_16x16x32_bf16(
              false, af[mi].v, false, bfr[ni].v, (short)0, acc[mi][ni], false, false);
    }
    async_wait();        // next tile fully in LDS
    __syncthreads();     // everyone done reading cur
    cur ^= 1;
  }

  // ---- epilogue: C/D layout (VGPR v -> M = rb+v, col = lane&15) ---------
  const int cbl = lane & 15;
  const int rb  = (lane < 16) ? 0 : 8;
#pragma unroll
  for (int mi = 0; mi < 2; ++mi) {
#pragma unroll
    for (int ni = 0; ni < 2; ++ni) {
      const int col = n0 + wn * 32 + ni * 16 + cbl;
      if (col >= N) continue;
      const float bsv = bias ? bias[col] : 0.f;
#pragma unroll
      for (int vv = 0; vv < 8; ++vv) {
        const int row = m0 + wm * 32 + mi * 16 + rb + vv;
        float val = acc[mi][ni][vv] + bsv;
        const long cidx = coff + (long)row * ldc + col;
        if constexpr (EPI == 0) {
          Cf[cidx] = val;
        } else if constexpr (EPI == 1) {
          Cb[cidx] = f2bf(val * scale);
        } else if constexpr (EPI == 2) {
          val = 0.5f * val * (1.0f + erff(val * 0.70710678118f));  // exact GELU
          Cb[cidx] = f2bf(val);
        } else if constexpr (EPI == 3) {
          Cf[cidx] = val + residual[(long)row * ldres + col];
        } else {  // EPI == 4: V stored as [B][D][T] for contiguous-K ctx GEMM
          Cb[(long)(row >> 10) * ((long)GD * GT) + (long)col * GT + (row & (GT - 1))] = f2bf(val);
        }
      }
    }
  }
}

// ---------------------------------------------------------------------------
// Transpose + fp32->bf16: src fp32 [R][C] (batched z), dst bf16 [C][R]
// ---------------------------------------------------------------------------
__global__ __launch_bounds__(256) void transpose_conv_kernel(
    const float* __restrict__ src, u16* __restrict__ dst, int R, int C)
{
  __shared__ float tile[32][33];
  const long zoff = (long)blockIdx.z * R * C;
  const int c0 = blockIdx.x * 32, r0 = blockIdx.y * 32;
  const int tx = threadIdx.x & 31, ty = threadIdx.x >> 5;
#pragma unroll
  for (int i = 0; i < 4; ++i) {
    int r = r0 + ty + i * 8, c = c0 + tx;
    float v = 0.f;
    if (r < R && c < C) v = src[zoff + (long)r * C + c];
    tile[ty + i * 8][tx] = v;
  }
  __syncthreads();
#pragma unroll
  for (int i = 0; i < 4; ++i) {
    int c = c0 + ty + i * 8, r = r0 + tx;
    if (c < C && r < R) dst[zoff + (long)c * R + r] = f2bf(tile[tx][ty + i * 8]);
  }
}

// ---------------------------------------------------------------------------
// Embedding: x[b,t,:] = tok_emb[inputs[b,t]] + pos_emb[t]
// ---------------------------------------------------------------------------
__global__ __launch_bounds__(256) void embed_kernel(
    const int* __restrict__ inp, const float* __restrict__ tok,
    const float* __restrict__ pos, float* __restrict__ x, int T_, int D_)
{
  int r = blockIdx.x;
  int t = r % T_;
  long id = inp[r];
  for (int i = threadIdx.x; i < D_; i += 256)
    x[(long)r * D_ + i] = tok[id * D_ + i] + pos[(long)t * D_ + i];
}

// ---------------------------------------------------------------------------
// LayerNorm (block per row) -> bf16 output
// ---------------------------------------------------------------------------
__global__ __launch_bounds__(256) void ln_kernel(
    const float* __restrict__ x, const float* __restrict__ g,
    const float* __restrict__ b, u16* __restrict__ out, int D_)
{
  __shared__ float r1[256], r2[256];
  long base = (long)blockIdx.x * D_;
  int tid = threadIdx.x;
  float s = 0.f, s2 = 0.f;
  float v[3];
#pragma unroll
  for (int i = 0; i < 3; ++i) {
    int idx = tid + i * 256;
    float f = 0.f;
    if (idx < D_) f = x[base + idx];
    v[i] = f; s += f; s2 += f * f;
  }
  r1[tid] = s; r2[tid] = s2; __syncthreads();
  for (int st = 128; st > 0; st >>= 1) {
    if (tid < st) { r1[tid] += r1[tid + st]; r2[tid] += r2[tid + st]; }
    __syncthreads();
  }
  float mean = r1[0] / D_;
  float var  = r2[0] / D_ - mean * mean;
  float rinv = rsqrtf(var + 1e-5f);
#pragma unroll
  for (int i = 0; i < 3; ++i) {
    int idx = tid + i * 256;
    if (idx < D_) out[base + idx] = f2bf((v[i] - mean) * rinv * g[idx] + b[idx]);
  }
}

// ---------------------------------------------------------------------------
// Causal softmax over one row of scores[z, t, :]; bf16 probs always,
// fp32 probs to attn (d_out) when non-null (last layer).
// ---------------------------------------------------------------------------
__global__ __launch_bounds__(256) void softmax_kernel(
    const float* __restrict__ scores, u16* __restrict__ probs,
    float* __restrict__ attn, int T_)
{
  __shared__ float red[256];
  int t = blockIdx.x;
  long base = ((long)blockIdx.y * T_ + t) * (long)T_;
  int tid = threadIdx.x;
  float vals[4];
  float vmax = -1e30f;
#pragma unroll
  for (int i = 0; i < 4; ++i) {
    int j = tid + i * 256;
    float sv = -1e30f;
    if (j < T_ && j <= t) sv = scores[base + j];
    vals[i] = sv;
    vmax = fmaxf(vmax, sv);
  }
  red[tid] = vmax; __syncthreads();
  for (int st = 128; st > 0; st >>= 1) {
    if (tid < st) red[tid] = fmaxf(red[tid], red[tid + st]);
    __syncthreads();
  }
  float m = red[0]; __syncthreads();
  float ssum = 0.f;
#pragma unroll
  for (int i = 0; i < 4; ++i) { vals[i] = expf(vals[i] - m); ssum += vals[i]; }
  red[tid] = ssum; __syncthreads();
  for (int st = 128; st > 0; st >>= 1) {
    if (tid < st) red[tid] += red[tid + st];
    __syncthreads();
  }
  float inv = 1.f / red[0];
#pragma unroll
  for (int i = 0; i < 4; ++i) {
    int j = tid + i * 256;
    if (j < T_) {
      float p = vals[i] * inv;
      probs[base + j] = f2bf(p);
      if (attn) attn[base + j] = p;
    }
  }
}

// ---------------------------------------------------------------------------
extern "C" void kernel_launch(void* const* d_in, const int* in_sizes, int n_in,
                              void* d_out, int out_size, void* d_ws, size_t ws_size,
                              hipStream_t stream)
{
  (void)in_sizes; (void)n_in; (void)out_size; (void)ws_size;
  const int*   inputs  = (const int*)  d_in[0];
  const float* tok_emb = (const float*)d_in[1];
  const float* pos_emb = (const float*)d_in[2];
  const float* ln1_g   = (const float*)d_in[3];
  const float* ln1_b   = (const float*)d_in[4];
  const float* wq      = (const float*)d_in[5];
  const float* bq      = (const float*)d_in[6];
  const float* wk      = (const float*)d_in[7];
  const float* bk      = (const float*)d_in[8];
  const float* wv      = (const float*)d_in[9];
  const float* bv      = (const float*)d_in[10];
  const float* wo      = (const float*)d_in[11];
  const float* bo      = (const float*)d_in[12];
  const float* ln2_g   = (const float*)d_in[13];
  const float* ln2_b   = (const float*)d_in[14];
  const float* w1      = (const float*)d_in[15];
  const float* b1      = (const float*)d_in[16];
  const float* w2      = (const float*)d_in[17];
  const float* b2      = (const float*)d_in[18];
  const float* lnf_g   = (const float*)d_in[19];
  const float* lnf_b   = (const float*)d_in[20];
  const float* proj_w  = (const float*)d_in[21];

  float* logits   = (float*)d_out;                       // [B,T,V]
  float* attn_out = logits + (size_t)GB * GT * GV;       // [B,H,T,T]

  char* ws = (char*)d_ws;
  size_t off = 0;
  auto alloc = [&](size_t bytes) -> void* {
    void* p = ws + off;
    off += (bytes + 255) & ~(size_t)255;
    return p;
  };
  float* x      = (float*)alloc((size_t)NT * GD * 4);              // residual stream
  u16*   hbuf   = (u16*)  alloc((size_t)NT * GD * 2);              // LN outputs (bf16)
  u16*   qb     = (u16*)  alloc((size_t)NT * GD * 2);              // [B*T][D]
  u16*   kb     = (u16*)  alloc((size_t)NT * GD * 2);              // [B*T][D]
  u16*   vT     = (u16*)  alloc((size_t)NT * GD * 2);              // [B][D][T]
  float* scores = (float*)alloc((size_t)GB * GH * GT * GT * 4);    // ~101 MB
  u16*   probs  = (u16*)  alloc((size_t)GB * GH * GT * GT * 2);    // ~50 MB
  u16*   ctxb   = (u16*)  alloc((size_t)NT * GD * 2);
  u16*   ffa    = (u16*)  alloc((size_t)NT * GFF * 2);
  // pre-transposed bf16 weights, all [N][K]
  u16*   wqT    = (u16*)  alloc((size_t)GL * GD * GD * 2);
  u16*   wkT    = (u16*)  alloc((size_t)GL * GD * GD * 2);
  u16*   wvT    = (u16*)  alloc((size_t)GL * GD * GD * 2);
  u16*   woT    = (u16*)  alloc((size_t)GL * GD * GD * 2);
  u16*   w1T    = (u16*)  alloc((size_t)GL * GD * GFF * 2);        // [FF][D] per layer
  u16*   w2T    = (u16*)  alloc((size_t)GL * GD * GFF * 2);        // [D][FF] per layer
  u16*   projT  = (u16*)  alloc((size_t)GV * GD * 2);              // [V][D]

  // weights -> bf16 [N][K] once per call
  dim3 gtw((GD + 31) / 32, (GD + 31) / 32, GL);
  transpose_conv_kernel<<<gtw, 256, 0, stream>>>(wq, wqT, GD, GD);
  transpose_conv_kernel<<<gtw, 256, 0, stream>>>(wk, wkT, GD, GD);
  transpose_conv_kernel<<<gtw, 256, 0, stream>>>(wv, wvT, GD, GD);
  transpose_conv_kernel<<<gtw, 256, 0, stream>>>(wo, woT, GD, GD);
  dim3 gt1((GFF + 31) / 32, (GD + 31) / 32, GL);
  transpose_conv_kernel<<<gt1, 256, 0, stream>>>(w1, w1T, GD, GFF);
  dim3 gt2((GD + 31) / 32, (GFF + 31) / 32, GL);
  transpose_conv_kernel<<<gt2, 256, 0, stream>>>(w2, w2T, GFF, GD);
  dim3 gt3((GV + 31) / 32, (GD + 31) / 32, 1);
  transpose_conv_kernel<<<gt3, 256, 0, stream>>>(proj_w, projT, GD, GV);

  embed_kernel<<<dim3(NT), 256, 0, stream>>>(inputs, tok_emb, pos_emb, x, GT, GD);

  const dim3 gDD(GD / TN, NT / TM, 1);          // [2048 x 768] outputs
  for (int l = 0; l < GL; ++l) {
    ln_kernel<<<dim3(NT), 256, 0, stream>>>(x, ln1_g + (size_t)l * GD, ln1_b + (size_t)l * GD, hbuf, GD);

    // QKV projections (q pre-scaled by 1/sqrt(DK); V stored [B][D][T])
    wmma_gemm<1><<<gDD, 256, 0, stream>>>(
        hbuf, GD, 0, 0, wqT + (size_t)l * GD * GD, GD, 0, 0, NT, GD, GD,
        bq + (size_t)l * GD, nullptr, 0, nullptr, qb, GD, 0, 0, 0.125f, 1);
    wmma_gemm<1><<<gDD, 256, 0, stream>>>(
        hbuf, GD, 0, 0, wkT + (size_t)l * GD * GD, GD, 0, 0, NT, GD, GD,
        bk + (size_t)l * GD, nullptr, 0, nullptr, kb, GD, 0, 0, 1.0f, 1);
    wmma_gemm<4><<<gDD, 256, 0, stream>>>(
        hbuf, GD, 0, 0, wvT + (size_t)l * GD * GD, GD, 0, 0, NT, GD, GD,
        bv + (size_t)l * GD, nullptr, 0, nullptr, vT, GD, 0, 0, 1.0f, 1);

    // scores[z] = q[z] * k[z]^T   (z = b*H+h; head slice via two-stride offsets)
    dim3 gsc(GT / TN, GT / TM, GB * GH);
    wmma_gemm<0><<<gsc, 256, 0, stream>>>(
        qb, GD, (long)GT * GD, GDK, kb, GD, (long)GT * GD, GDK, GT, GT, GDK,
        nullptr, nullptr, 0, scores, nullptr, GT, (long)GH * GT * GT, (long)GT * GT, 1.0f, GH);

    softmax_kernel<<<dim3(GT, GB * GH), 256, 0, stream>>>(
        scores, probs, (l == GL - 1) ? attn_out : nullptr, GT);

    // ctx[z] = probs[z] * v[z]  (B operand = vT slice [64][T], contiguous K)
    dim3 gctx(1, GT / TM, GB * GH);
    wmma_gemm<1><<<gctx, 256, 0, stream>>>(
        probs, GT, (long)GH * GT * GT, (long)GT * GT, vT, GT, (long)GD * GT, (long)GDK * GT,
        GT, GDK, GT, nullptr, nullptr, 0, nullptr, ctxb, GD, (long)GT * GD, GDK, 1.0f, GH);

    // x = x + ctx * wo + bo
    wmma_gemm<3><<<gDD, 256, 0, stream>>>(
        ctxb, GD, 0, 0, woT + (size_t)l * GD * GD, GD, 0, 0, NT, GD, GD,
        bo + (size_t)l * GD, x, GD, x, nullptr, GD, 0, 0, 1.0f, 1);

    ln_kernel<<<dim3(NT), 256, 0, stream>>>(x, ln2_g + (size_t)l * GD, ln2_b + (size_t)l * GD, hbuf, GD);

    // ffa = gelu(h2 * w1 + b1)
    dim3 gf1(GFF / TN, NT / TM, 1);
    wmma_gemm<2><<<gf1, 256, 0, stream>>>(
        hbuf, GD, 0, 0, w1T + (size_t)l * GD * GFF, GD, 0, 0, NT, GFF, GD,
        b1 + (size_t)l * GFF, nullptr, 0, nullptr, ffa, GFF, 0, 0, 1.0f, 1);

    // x = x + ffa * w2 + b2
    wmma_gemm<3><<<gDD, 256, 0, stream>>>(
        ffa, GFF, 0, 0, w2T + (size_t)l * GD * GFF, GFF, 0, 0, NT, GD, GFF,
        b2 + (size_t)l * GD, x, GD, x, nullptr, GD, 0, 0, 1.0f, 1);
  }

  // final LN + LM head (ragged N handled by clamp + store guard)
  ln_kernel<<<dim3(NT), 256, 0, stream>>>(x, lnf_g, lnf_b, hbuf, GD);
  dim3 glog((GV + TN - 1) / TN, NT / TM, 1);
  wmma_gemm<0><<<glog, 256, 0, stream>>>(
      hbuf, GD, 0, 0, projT, GD, 0, 0, NT, GV, GD,
      nullptr, nullptr, 0, logits, nullptr, GV, 0, 0, 1.0f, 1);
}
